// Convolution2DColorEnergy_DelayBasis6_88751204205053
// MI455X (gfx1250) — matrix operand order repaired and checked by reading, not compiled, not run on hardware
//
#include <hip/hip_runtime.h>

typedef _Float16 h16;
typedef __attribute__((ext_vector_type(16))) _Float16 v16h;
typedef __attribute__((ext_vector_type(8)))  float    v8f;
typedef __attribute__((__vector_size__(4 * sizeof(int)))) int v4i_t;

#define B_   8
#define CIN  3
#define DD   12
#define TT   6
#define HH   128
#define WW   128
#define HO   122
#define WO   122
#define CO   34
#define KTOT 147          // CIN*7*7
#define KPAD 160          // padded to 5 x 32
#define NTI  3            // N tiles of 16 (34 -> 48)
#define KS   5            // K steps of 32

// workspace layout (bytes)
#define XT_BYTES  (B_*TT*CIN*HH*WW*2)       // 4,718,592  (f16 projected input)
#define BF_OFF    XT_BYTES                   // 15,360     (packed WMMA B frags)
#define CO1_OFF   (XT_BYTES + 0x4000)        // f16 conv output, 48.6 MB
// total ws use ~53.3 MB

// ---------------------------------------------------------------------------
// CDNA5 async global->LDS DMA (ASYNCcnt path), with compile-safe fallback.
// Builtin signature (from compiler diagnostic): (v4i32 AS1*, v4i32 AS3*, imm, imm)
// ---------------------------------------------------------------------------
#if __has_builtin(__builtin_amdgcn_global_load_async_to_lds_b128)
#define ASYNC_LDS_OK 1
#else
#define ASYNC_LDS_OK 0
#endif

__device__ __forceinline__ void copy16_g2l(const void* g, void* l) {
#if ASYNC_LDS_OK
    __builtin_amdgcn_global_load_async_to_lds_b128(
        (__attribute__((address_space(1))) v4i_t*)g,
        (__attribute__((address_space(3))) v4i_t*)l, 0, 0);
#else
    *(uint4*)l = *(const uint4*)g;
#endif
}

__device__ __forceinline__ void async_join() {
#if ASYNC_LDS_OK
#if __has_builtin(__builtin_amdgcn_s_wait_asynccnt)
    __builtin_amdgcn_s_wait_asynccnt(0);
#else
    asm volatile("s_wait_asynccnt 0x0" ::: "memory");
#endif
#endif
}

// ---------------------------------------------------------------------------
// Kernel 0: pack conv weights into WMMA B-fragment lanes, contiguous k order (f16).
// B is KxN (32x16) per fragment: lanes 0-15 hold K=0..15 of column n (elem j
// <-> K=j), lanes 16-31 hold K=16..31. Zero-pad K>=147 and O>=34.
// ---------------------------------------------------------------------------
__global__ void k_pack_w(const float* __restrict__ W, h16* __restrict__ bfrag) {
    int task = blockIdx.x * blockDim.x + threadIdx.x;   // (ks*NTI + nt)*32 + lane
    if (task >= KS * NTI * 32) return;
    int ks  = task / (NTI * 32);
    int rem = task - ks * (NTI * 32);
    int nt  = rem >> 5;
    int l   = rem & 31;
    int n   = l & 15, hi = l >> 4;
    int o   = nt * 16 + n;
    h16* dst = bfrag + (size_t)task * 16;
    #pragma unroll
    for (int j = 0; j < 16; ++j) {
        int k = ks * 32 + hi * 16 + j;
        float v = 0.f;
        if (k < KTOT && o < CO) v = W[o * KTOT + k];    // W flat (34,3,1,7,7)
        dst[j] = (h16)v;
    }
}

// ---------------------------------------------------------------------------
// Kernel 1: temporal basis projection x(b,c,d,h,w) * Wt(d,t) -> xt f16,
// layout xt[(b*6+t)][c][h][w]. Commutes with the conv since kd==1, and
// halves the conv FLOPs (D=12 -> T=6 slices).
// ---------------------------------------------------------------------------
__global__ void k_temporal(const float* __restrict__ x,
                           const float* __restrict__ Wt,
                           h16* __restrict__ xt) {
    __shared__ float wts[DD * TT];
    int tid = threadIdx.x;
    if (tid < DD * TT) wts[tid] = Wt[tid];
    __syncthreads();
    int idx = blockIdx.x * blockDim.x + tid;            // over B*CIN*HH*WW (exact)
    int hw  = idx & (HH * WW - 1);
    int r   = idx >> 14;
    int c   = r % CIN;
    int b   = r / CIN;
    float acc[TT] = {0.f, 0.f, 0.f, 0.f, 0.f, 0.f};
    const float* xp = x + ((size_t)(b * CIN + c) * DD) * (HH * WW) + hw;
    #pragma unroll
    for (int d = 0; d < DD; ++d) {
        float xv = xp[d * HH * WW];
        #pragma unroll
        for (int t = 0; t < TT; ++t) acc[t] += xv * wts[d * TT + t];
    }
    #pragma unroll
    for (int t = 0; t < TT; ++t)
        xt[((size_t)((b * TT + t) * CIN + c)) * (HH * WW) + hw] = (h16)acc[t];
}

// ---------------------------------------------------------------------------
// Kernel 2: implicit-GEMM 7x7 conv via v_wmma_f32_16x16x32_f16.
// Workgroup = 8 waves = 8 output rows x 16 cols of one (b,t) slice.
// Halo patch (3ch x 14 rows x 24 cols, f16) DMA'd to LDS with async b128
// tiles (no clamping: out-of-slice reads stay inside the workspace and only
// feed lanes whose stores are masked off).
// ---------------------------------------------------------------------------
#define ROWS   8
#define PPITCH 24
#define PROWS  (ROWS + 6)                  // 14
#define PSIZE  (CIN * PROWS * PPITCH)      // 1008 halves
#define PTOT   (PSIZE + 256)               // + zeroed pad for K >= 147

__global__ void __launch_bounds__(256) k_conv(const h16* __restrict__ xt,
                                              const h16* __restrict__ bfrag,
                                              h16* __restrict__ co1) {
    __shared__ __align__(16) h16 patch[PTOT];
    __shared__ __align__(16) int atbl[KS][2][16];   // byte offsets, per (ks, lane-half)
    int tid = threadIdx.x;
    int w0  = blockIdx.x * 16;
    int h0  = blockIdx.y * ROWS;
    int bt  = blockIdx.z;

    // 1) async-DMA halo patch: 3ch x 14 rows x 3 b128 tiles (rows are 48B)
    for (int t = tid; t < CIN * PROWS * 3; t += 256) {
        int c  = t / (PROWS * 3);
        int rr = t - c * (PROWS * 3);
        int r  = rr / 3;
        int g  = (rr - r * 3) * 8;
        const h16* gp = xt + ((size_t)(bt * CIN + c) * HH + (h0 + r)) * WW + w0 + g;
        copy16_g2l(gp, &patch[(c * PROWS + r) * PPITCH + g]);
    }
    // 2) zero the K-padding target region (vector stores)
    if (tid < 64) {
        uint2 z; z.x = 0u; z.y = 0u;
        *(uint2*)&patch[PSIZE + tid * 4] = z;
    }
    // 3) pre-scaled A-gather byte-offset table, contiguous per (ks, lane-half)
    //    A k order used here (same as B packing): K = hi*16 + j
    for (int t = tid; t < KS * 32; t += 256) {
        int ks = t >> 5, l = t & 31, hi = l >> 4, j = l & 15;
        int kk = ks * 32 + hi * 16 + j;
        int off = 2 * PSIZE;                           // -> zeroed pad
        if (kk < KTOT) {
            int c = kk / 49, r = kk - c * 49, kh = r / 7, kw = r - kh * 7;
            off = 2 * ((c * PROWS + kh) * PPITCH + kw);
        }
        atbl[ks][hi][j] = off;
    }
    async_join();
    __syncthreads();

    int wave = tid >> 5;
    int lane = tid & 31;
    int m    = lane & 15;
    int hi   = lane >> 4;
    const char* pl = (const char*)patch + 2 * (wave * PPITCH + m);

    v8f acc0 = {0.f,0.f,0.f,0.f,0.f,0.f,0.f,0.f};
    v8f acc1 = {0.f,0.f,0.f,0.f,0.f,0.f,0.f,0.f};
    v8f acc2 = {0.f,0.f,0.f,0.f,0.f,0.f,0.f,0.f};
    const v16h* bf = (const v16h*)bfrag;

    #pragma unroll
    for (int ks = 0; ks < KS; ++ks) {
        int offs[16];
        *(int4*)&offs[0]  = *(const int4*)&atbl[ks][hi][0];
        *(int4*)&offs[4]  = *(const int4*)&atbl[ks][hi][4];
        *(int4*)&offs[8]  = *(const int4*)&atbl[ks][hi][8];
        *(int4*)&offs[12] = *(const int4*)&atbl[ks][hi][12];
        v16h a;
        #pragma unroll
        for (int j = 0; j < 16; ++j)
            a[j] = *(const h16*)(pl + offs[j]);

        v16h b0 = bf[(ks * NTI + 0) * 32 + lane];
        acc0 = __builtin_amdgcn_wmma_f32_16x16x32_f16(false, a, false, b0,
                                                      (short)0, acc0, false, false);
        v16h b1 = bf[(ks * NTI + 1) * 32 + lane];
        acc1 = __builtin_amdgcn_wmma_f32_16x16x32_f16(false, a, false, b1,
                                                      (short)0, acc1, false, false);
        v16h b2 = bf[(ks * NTI + 2) * 32 + lane];
        acc2 = __builtin_amdgcn_wmma_f32_16x16x32_f16(false, a, false, b2,
                                                      (short)0, acc2, false, false);
    }

    // D layout: VGPR j -> M = j + hi*8, lane&15 -> N. Channel-contiguous store.
    int h = h0 + wave;
    if (h < HO) {
        #pragma unroll
        for (int j = 0; j < 8; ++j) {
            int wp = w0 + hi * 8 + j;
            if (wp < WO) {
                size_t base = ((size_t)(bt * HO + h) * WO + wp) * CO;
                co1[base + m]      = (h16)acc0[j];
                co1[base + 16 + m] = (h16)acc1[j];
                if (m < 2) co1[base + 32 + m] = (h16)acc2[j];   // o = 32,33
            }
        }
    }
}

// ---------------------------------------------------------------------------
// Kernel 3: per-pixel 12x12 einsums, energy sqrt, bias. One thread per
// (b, f in 0..16, h, w). Wm async-staged in LDS. Bandwidth-bound.
// ---------------------------------------------------------------------------
__global__ void k_energy(const h16* __restrict__ co1,
                         const float* __restrict__ Wm,
                         const float* __restrict__ bias,
                         float* __restrict__ out) {
    __shared__ __align__(16) float wm[17 * 12 * 24];
    int tid = threadIdx.x;
    for (int t = tid; t < (17 * 12 * 24) / 4; t += 256)   // 1224 b128 tiles
        copy16_g2l(Wm + t * 4, &wm[t * 4]);
    async_join();
    __syncthreads();

    int idx = blockIdx.x * 256 + tid;
    const int TOTAL = B_ * 17 * HO * WO;
    if (idx >= TOTAL) return;
    int w = idx % WO;
    int r = idx / WO;
    int h = r % HO; r /= HO;
    int f = r % 17;
    int b = r / 17;

    int olo = (f < 16) ? f : 16;        // c12/c34 channel mapping
    int ohi = (f < 16) ? f + 17 : 33;
    float v[12];
    #pragma unroll
    for (int t6 = 0; t6 < TT; ++t6) {
        size_t base = ((size_t)((b * TT + t6) * HO + h) * WO + w) * CO;
        v[t6]     = (float)co1[base + olo];
        v[6 + t6] = (float)co1[base + ohi];
    }
    float bs = bias[f];
    const float* wrow = &wm[f * 12 * 24];
    #pragma unroll
    for (int t = 0; t < 12; ++t) {
        float e1 = 0.f, e2 = 0.f;
        #pragma unroll
        for (int k = 0; k < 12; ++k) {
            float vk = v[k];
            e1 += vk * wrow[k * 24 + t];
            e2 += vk * wrow[k * 24 + 12 + t];
        }
        // f<16: energy = sqrt(e1^2+e2^2+eps); f==16: lin = both halves summed
        float res = (f < 16) ? sqrtf(e1 * e1 + e2 * e2 + 1e-7f) : (e1 + e2);
        out[(((size_t)(b * 12 + t) * 17 + f) * HO + h) * WO + w] = res + bs;
    }
}

// ---------------------------------------------------------------------------
extern "C" void kernel_launch(void* const* d_in, const int* in_sizes, int n_in,
                              void* d_out, int out_size, void* d_ws, size_t ws_size,
                              hipStream_t stream) {
    const float* x  = (const float*)d_in[0];  // (8,3,12,128,128)
    const float* W  = (const float*)d_in[1];  // (34,3,1,7,7)
    const float* Wt = (const float*)d_in[2];  // (12,6)
    const float* Wm = (const float*)d_in[3];  // (17,12,24)
    const float* bv = (const float*)d_in[4];  // (17,)
    float* out = (float*)d_out;               // (8,12,17,122,122)

    char* ws = (char*)d_ws;
    h16* xt    = (h16*)(ws);                  // f16 projected input
    h16* bfrag = (h16*)(ws + BF_OFF);         // packed WMMA B fragments
    h16* co1   = (h16*)(ws + CO1_OFF);        // f16 conv+projection output

    k_pack_w<<<2, 256, 0, stream>>>(W, bfrag);
    k_temporal<<<(B_ * CIN * HH * WW) / 256, 256, 0, stream>>>(x, Wt, xt);
    dim3 g2((WO + 15) / 16, (HO + ROWS - 1) / ROWS, B_ * TT);   // 8 x 16 x 48
    k_conv<<<g2, 256, 0, stream>>>(xt, bfrag, co1);
    int n3 = B_ * 17 * HO * WO;
    k_energy<<<(n3 + 255) / 256, 256, 0, stream>>>(co1, Wm, bv, out);
}
